// RAFTMVS_2Layer_90288802497366
// MI455X (gfx1250) — matrix operand order repaired and checked by reading, not compiled, not run on hardware
//
#include <hip/hip_runtime.h>
#include <hip/hip_bf16.h>
#include <cstdint>
#include <cstddef>

// ---------------------------------------------------------------------------
// RAFT-MVS 2-level refiner for gfx1250 (MI455X).
// Conv layers -> implicit GEMM via V_WMMA_F32_16X16X32_F16 (wave32 WMMA),
// B operand staged through LDS with GLOBAL_LOAD_ASYNC_TO_LDS_B128 (ASYNCcnt),
// double-buffered, 4-way N register blocking per wave.
// ---------------------------------------------------------------------------

typedef __attribute__((ext_vector_type(16))) _Float16 v16h;
typedef __attribute__((ext_vector_type(8)))  float    v8f;
typedef _Float16 half_t;

constexpr int H   = 48,  W  = 64,  HW  = H * W;       // 1/4-res feature grid
constexpr int H2  = 24,  W2 = 32,  HW2 = H2 * W2;     // 1/8-res GRU grid
constexpr int HID = 128;
constexpr int D   = 64;                                // depth samples
constexpr int HI  = 192, WI = 256;                     // full-res output
constexpr int CORP = 36;                               // corr planes

__device__ __forceinline__ float sigm(float x) { return 1.f / (1.f + expf(-x)); }

// ---------------------------------------------------------------------------
// elementwise utilities
// ---------------------------------------------------------------------------
__global__ void zero_k(float* __restrict__ d, int n) {
    int i = blockIdx.x * blockDim.x + threadIdx.x;
    if (i < n) d[i] = 0.f;
}
__global__ void copy_k(float* __restrict__ d, const float* __restrict__ s, int n) {
    int i = blockIdx.x * blockDim.x + threadIdx.x;
    if (i < n) d[i] = s[i];
}
__global__ void tanh_k(float* __restrict__ d, const float* __restrict__ s, int n) {
    int i = blockIdx.x * blockDim.x + threadIdx.x;
    if (i < n) d[i] = tanhf(s[i]);
}
__global__ void add_inplace_k(float* __restrict__ d, const float* __restrict__ a, int n) {
    int i = blockIdx.x * blockDim.x + threadIdx.x;
    if (i < n) d[i] += a[i];
}
// rh = sigmoid(r_pre + cr) * h
__global__ void gru_rh_k(const float* __restrict__ rp, const float* __restrict__ cr,
                         const float* __restrict__ h, float* __restrict__ rh, int n) {
    int i = blockIdx.x * blockDim.x + threadIdx.x;
    if (i < n) rh[i] = sigm(rp[i] + cr[i]) * h[i];
}
// hn = (1-z)*h + z*q,  z = sigmoid(zp+cz), q = tanh(qp+cq)
__global__ void gru_comb_k(const float* __restrict__ zp, const float* __restrict__ cz,
                           const float* __restrict__ qp, const float* __restrict__ cq,
                           const float* __restrict__ h, float* __restrict__ hn, int n) {
    int i = blockIdx.x * blockDim.x + threadIdx.x;
    if (i < n) {
        float z = sigm(zp[i] + cz[i]);
        float q = tanhf(qp[i] + cq[i]);
        hn[i] = (1.f - z) * h[i] + z * q;
    }
}

// ---------------------------------------------------------------------------
// pool2x: 3x3 avg, stride 2, pad 1 :  (HID,H,W) -> (HID,H2,W2), sum/9
// ---------------------------------------------------------------------------
__global__ void pool2x_k(const float* __restrict__ in, float* __restrict__ out, int n) {
    int i = blockIdx.x * blockDim.x + threadIdx.x;
    if (i >= n) return;
    int c = i / HW2, r = i % HW2;
    int oy = r / W2, ox = r % W2;
    float s = 0.f;
    for (int dy = -1; dy <= 1; ++dy)
        for (int dx = -1; dx <= 1; ++dx) {
            int y = 2 * oy + dy, x = 2 * ox + dx;
            if (y >= 0 && y < H && x >= 0 && x < W) s += in[(c * H + y) * W + x];
        }
    out[i] = s / 9.f;
}

// ---------------------------------------------------------------------------
// bilinear up (align-corners): (HID,H2,W2) -> (HID,H,W)
// ---------------------------------------------------------------------------
__global__ void interp_up_k(const float* __restrict__ in, float* __restrict__ out, int n) {
    int i = blockIdx.x * blockDim.x + threadIdx.x;
    if (i >= n) return;
    int c = i / HW, r = i % HW;
    int oy = r / W, ox = r % W;
    float yi = oy * (float)(H2 - 1) / (float)(H - 1);
    float xi = ox * (float)(W2 - 1) / (float)(W - 1);
    int y0 = (int)floorf(yi); int y1 = min(y0 + 1, H2 - 1);
    int x0 = (int)floorf(xi); int x1 = min(x0 + 1, W2 - 1);
    float wy = yi - y0, wx = xi - x0;
    const float* p = in + c * HW2;
    float t0 = p[y0 * W2 + x0] * (1.f - wy) + p[y1 * W2 + x0] * wy;
    float t1 = p[y0 * W2 + x1] * (1.f - wy) + p[y1 * W2 + x1] * wy;
    out[i] = t0 * (1.f - wx) + t1 * wx;
}

// ---------------------------------------------------------------------------
// projection setup: scale rows 0..1 by 0.25, invert ref (Gauss-Jordan),
// proj_s = src_s @ inv(ref); emit rot(9)+trans(3) per source (2 sources)
// ---------------------------------------------------------------------------
__global__ void compute_proj_k(const float* __restrict__ P, float* __restrict__ out) {
    if (threadIdx.x != 0 || blockIdx.x != 0) return;
    float aug[4][8];
    for (int r = 0; r < 4; ++r)
        for (int c = 0; c < 4; ++c) {
            float v = P[r * 4 + c];
            if (r < 2) v *= 0.25f;
            aug[r][c] = v;
            aug[r][4 + c] = (r == c) ? 1.f : 0.f;
        }
    for (int i = 0; i < 4; ++i) {
        float inv = 1.f / aug[i][i];
        for (int c = 0; c < 8; ++c) aug[i][c] *= inv;
        for (int r = 0; r < 4; ++r)
            if (r != i) {
                float f = aug[r][i];
                for (int c = 0; c < 8; ++c) aug[r][c] -= f * aug[i][c];
            }
    }
    for (int s = 0; s < 2; ++s) {
        const float* Sp = P + 16 * (s + 1);
        float* o = out + s * 12;
        for (int r = 0; r < 3; ++r) {
            float pr[4];
            for (int c = 0; c < 4; ++c) {
                float acc = 0.f;
                for (int k = 0; k < 4; ++k) {
                    float sv = Sp[r * 4 + k];
                    if (r < 2) sv *= 0.25f;
                    acc += sv * aug[k][4 + c];
                }
                pr[c] = acc;
            }
            o[r * 3 + 0] = pr[0]; o[r * 3 + 1] = pr[1]; o[r * 3 + 2] = pr[2];
            o[9 + r] = pr[3];
        }
    }
}

// ---------------------------------------------------------------------------
// warp + correlation volume: one block per pixel, 256 threads = channels,
// loop over 64 depths; LDS tree reduce the channel dot product.
// corr0 layout: [pix][d]
// ---------------------------------------------------------------------------
__global__ __launch_bounds__(256)
void warp_corr_k(const float* __restrict__ f1, const float* __restrict__ f2,
                 const float* __restrict__ pj, const float* __restrict__ dmin,
                 const float* __restrict__ dmax, float* __restrict__ corr0) {
    __shared__ float red[256];
    int pix = blockIdx.x;
    int c = threadIdx.x;
    int y = pix / W, x = pix % W;
    float inv_min = 1.f / dmin[0], inv_max = 1.f / dmax[0];
    float rx[2], ry[2], rz[2], t0[2], t1[2], t2[2];
    for (int s = 0; s < 2; ++s) {
        const float* pr = pj + s * 12;
        rx[s] = pr[0] * x + pr[1] * y + pr[2];
        ry[s] = pr[3] * x + pr[4] * y + pr[5];
        rz[s] = pr[6] * x + pr[7] * y + pr[8];
        t0[s] = pr[9]; t1[s] = pr[10]; t2[s] = pr[11];
    }
    for (int d = 0; d < D; ++d) {
        float depth = 1.f / (inv_max + ((float)d / (D - 1)) * (inv_min - inv_max));
        float wv[2];
        for (int s = 0; s < 2; ++s) {
            float px = rx[s] * depth + t0[s];
            float py = ry[s] * depth + t1[s];
            float pz = rz[s] * depth + t2[s];
            if (pz <= 0.001f) { px = (float)W; py = (float)H; pz = 1.f; }
            float ix = px / pz, iy = py / pz;
            float x0 = floorf(ix), y0 = floorf(iy);
            float wx = ix - x0, wy = iy - y0;
            const float* img = ((s == 0) ? f1 : f2) + (size_t)c * HW;
            auto tap = [&](float xf, float yf, float wgt) -> float {
                bool valid = (xf >= 0.f) && (xf <= (float)(W - 1)) &&
                             (yf >= 0.f) && (yf <= (float)(H - 1));
                int xi = min(max((int)xf, 0), W - 1);
                int yi = min(max((int)yf, 0), H - 1);
                return valid ? img[yi * W + xi] * wgt : 0.f;
            };
            wv[s] = tap(x0, y0, (1.f - wx) * (1.f - wy)) +
                    tap(x0 + 1.f, y0, wx * (1.f - wy)) +
                    tap(x0, y0 + 1.f, (1.f - wx) * wy) +
                    tap(x0 + 1.f, y0 + 1.f, wx * wy);
        }
        red[c] = wv[0] * wv[1];
        __syncthreads();
        for (int st = 128; st > 0; st >>= 1) {
            if (c < st) red[c] += red[c + st];
            __syncthreads();
        }
        if (c == 0) corr0[(size_t)pix * D + d] = red[0] / (256.f * 8.f);
        __syncthreads();
    }
}

// pyramid level 1: average depth pairs
__global__ void pyr1_k(const float* __restrict__ c0, float* __restrict__ c1, int n) {
    int i = blockIdx.x * blockDim.x + threadIdx.x;
    if (i >= n) return;
    int pix = i / 32, d = i % 32;
    c1[i] = 0.5f * (c0[pix * D + 2 * d] + c0[pix * D + 2 * d + 1]);
}

// ---------------------------------------------------------------------------
// corr lookup: channels = [disp0: lvl0 9, lvl1 9 | disp1: lvl0 9, lvl1 9]
// ---------------------------------------------------------------------------
__global__ void corr_lookup_k(const float* __restrict__ disp,
                              const float* __restrict__ p0,
                              const float* __restrict__ p1,
                              float* __restrict__ cf) {
    int n = blockIdx.x * blockDim.x + threadIdx.x;
    if (n >= HW) return;
    for (int j = 0; j < 2; ++j) {
        float d0 = disp[j * HW + n];
        for (int lvl = 0; lvl < 2; ++lvl) {
            int dl = D >> lvl;
            const float* vol = (lvl == 0) ? (p0 + (size_t)n * D) : (p1 + (size_t)n * 32);
            float cbase = d0 / (float)(1 << lvl);
            for (int t = 0; t < 9; ++t) {
                float cc = cbase + (float)(t - 4);
                float x0 = floorf(cc);
                float wg = cc - x0;
                auto g = [&](float xf) -> float {
                    bool valid = (xf >= 0.f) && (xf <= (float)(dl - 1));
                    int xi = min(max((int)xf, 0), dl - 1);
                    return valid ? vol[xi] : 0.f;
                };
                float v = g(x0) * (1.f - wg) + g(x0 + 1.f) * wg;
                int ch = j * 18 + lvl * 9 + t;
                cf[(size_t)ch * HW + n] = v;
            }
        }
    }
}

// ---------------------------------------------------------------------------
// convex upsample (factor 4) + mask*0.25 softmax over 9 taps
// ---------------------------------------------------------------------------
__global__ void upsample_k(const float* __restrict__ flow, const float* __restrict__ mask,
                           float* __restrict__ out, int n) {
    int i = blockIdx.x * blockDim.x + threadIdx.x;
    if (i >= n) return;
    int ch = i / (HI * WI), r = i % (HI * WI);
    int oy = r / WI, ox = r % WI;
    int y = oy >> 2, fy = oy & 3, x = ox >> 2, fx = ox & 3;
    float l[9], mx = -1e30f;
    for (int k = 0; k < 9; ++k) {
        l[k] = 0.25f * mask[(size_t)((k << 4) + (fy << 2) + fx) * HW + y * W + x];
        mx = fmaxf(mx, l[k]);
    }
    float se = 0.f;
    for (int k = 0; k < 9; ++k) { l[k] = expf(l[k] - mx); se += l[k]; }
    float acc = 0.f;
    for (int di = 0; di < 3; ++di)
        for (int dj = 0; dj < 3; ++dj) {
            int yy = y + di - 1, xx = x + dj - 1;
            float f = (yy >= 0 && yy < H && xx >= 0 && xx < W)
                        ? flow[(size_t)ch * HW + yy * W + xx] : 0.f;
            acc += (l[di * 3 + dj] / se) * f;
        }
    out[i] = acc;
}

__global__ void depth_k(const float* __restrict__ du, const float* __restrict__ dmin,
                        const float* __restrict__ dmax, float* __restrict__ out, int n) {
    int i = blockIdx.x * blockDim.x + threadIdx.x;
    if (i >= n) return;
    float inv_min = 1.f / dmin[0], inv_max = 1.f / dmax[0];
    float inv_d = inv_max + fmaxf(du[i], 0.001f) * (inv_min - inv_max) / (float)(D - 1);
    out[i] = 1.f / inv_d;
}

// ---------------------------------------------------------------------------
// WMMA GEMM path.
// A fragment (16x32 f16): lane<16 holds row=lane, K=[0..7,16..23];
//                         lane>=16 holds row=lane-16, K=[8..15,24..31].
// B fragment (32x16 f16): lane<16 -> col=lane, K=[0..7,16..23];
//                         lane>=16 -> col=lane-16, K=[8..15,24..31].
// C/D (16x16 f32): VGPR v, lane<16 -> row v, col lane; lane>=16 -> row v+8.
// ---------------------------------------------------------------------------
__global__ void pack_w_k(const float* __restrict__ Wsrc, half_t* __restrict__ A,
                         int O, int K, int Ksteps, long total) {
    long idx = (long)blockIdx.x * blockDim.x + threadIdx.x;
    if (idx >= total) return;
    int e = idx & 15;
    int lane = (idx >> 4) & 31;
    long fs = idx >> 9;
    int ks = (int)(fs % Ksteps);
    int mtile = (int)(fs / Ksteps);
    int row = lane & 15, kg = lane >> 4;
    int klocal = (e < 8) ? (kg * 8 + e) : (16 + kg * 8 + (e - 8));
    int k = ks * 32 + klocal;
    int o = mtile * 16 + row;
    float v = (o < O && k < K) ? Wsrc[(size_t)o * K + k] : 0.f;
    A[idx] = (half_t)v;
}

__global__ void im2col_k(const float* __restrict__ src, half_t* __restrict__ B,
                         int C, int Hh, int Ww, int ksz, int pad, int Ksteps, long total) {
    long idx = (long)blockIdx.x * blockDim.x + threadIdx.x;
    if (idx >= total) return;
    int e = idx & 15;
    int lane = (idx >> 4) & 31;
    long fs = idx >> 9;
    int ks = (int)(fs % Ksteps);
    int ntile = (int)(fs / Ksteps);
    int col = lane & 15;
    int klocal = (e < 8) ? ((lane >> 4) * 8 + e) : (16 + (lane >> 4) * 8 + (e - 8));
    int k = ks * 32 + klocal;
    int n = ntile * 16 + col;
    int y = n / Ww, x = n % Ww;
    int K = C * ksz * ksz;
    float v = 0.f;
    if (k < K) {
        int c = k / (ksz * ksz), r = k % (ksz * ksz);
        int sy = y + r / ksz - pad, sx = x + r % ksz - pad;
        if (sy >= 0 && sy < Hh && sx >= 0 && sx < Ww)
            v = src[((size_t)c * Hh + sy) * Ww + sx];
    }
    B[idx] = (half_t)v;
}

// ---------------------------------------------------------------------------
// GEMM: block = 8 waves = 128(M) x 64(N) tile. B fragments (4 ntiles x 1KB
// per K step) staged into double-buffered LDS via GLOBAL_LOAD_ASYNC_TO_LDS,
// one b128 async copy per thread per K step; A streamed per-wave from global.
// 4 WMMAs per wave per K step.
// ---------------------------------------------------------------------------
__global__ __launch_bounds__(256)
void gemm_wmma_k(const half_t* __restrict__ A, const half_t* __restrict__ B,
                 const float* __restrict__ bias, float* __restrict__ out,
                 int M, int N, int Ksteps, int Mtiles, int act) {
    __shared__ __align__(128) half_t sB[2][4 * 512];   // 2 x 4KB stages

    const int tid  = threadIdx.x;
    const int lane = tid & 31;
    const int wave = tid >> 5;
    const int nblk = blockIdx.x;                 // 64-column block
    const int mtile = blockIdx.y * 8 + wave;
    const bool active = (mtile < Mtiles);        // wave-uniform

    // async-copy duty: thread t moves 16B of the 4KB stage
    const int nt_cp  = tid >> 6;                 // which ntile fragment
    const int foff   = (tid & 63) << 4;          // byte offset inside fragment
    const unsigned lds_base =
        (unsigned)(uintptr_t)(__attribute__((address_space(3))) half_t*)&sB[0][0];
    const unsigned lds_my = lds_base + ((unsigned)tid << 4);

    auto issue_stage = [&](int ks, int buf) {
        const char* src = (const char*)B +
            ((((size_t)(nblk * 4 + nt_cp)) * Ksteps + (size_t)ks) << 10) + foff;
        unsigned lds = lds_my + (unsigned)(buf << 12);
        unsigned long long ga = (unsigned long long)src;
        asm volatile("global_load_async_to_lds_b128 %0, %1, off"
                     :: "v"(lds), "v"(ga) : "memory");
    };

    const v16h* Af = (const v16h*)A + (size_t)mtile * Ksteps * 32 + lane;
    v8f acc[4] = {};

    issue_stage(0, 0);
    for (int ks = 0; ks < Ksteps; ++ks) {
        int buf = ks & 1;
        if (ks + 1 < Ksteps) {
            issue_stage(ks + 1, buf ^ 1);
            asm volatile("s_wait_asynccnt 0x1" ::: "memory");
        } else {
            asm volatile("s_wait_asynccnt 0x0" ::: "memory");
        }
        __syncthreads();                          // stage ks visible to all waves
        if (active) {
            if (ks + 2 < Ksteps)
                __builtin_prefetch((const void*)(Af + (size_t)(ks + 2) * 32), 0, 1);
            v16h av = Af[(size_t)ks * 32];
#pragma unroll
            for (int nt = 0; nt < 4; ++nt) {
                v16h bv = *(const v16h*)&sB[buf][nt * 512 + lane * 16];
                acc[nt] = __builtin_amdgcn_wmma_f32_16x16x32_f16(
                    false, av, false, bv, (short)0, acc[nt], false, false);
            }
        }
        __syncthreads();                          // all reads done before overwrite
    }

    if (active) {
        int col = lane & 15;
        int rbase = (lane >> 4) * 8;
#pragma unroll
        for (int nt = 0; nt < 4; ++nt) {
            int gn = (nblk * 4 + nt) * 16 + col;
#pragma unroll
            for (int v = 0; v < 8; ++v) {
                int gm = mtile * 16 + rbase + v;
                if (gm < M) {
                    float val = acc[nt][v] + bias[gm];
                    if (act == 1) val = fmaxf(val, 0.f);
                    out[(size_t)gm * N + gn] = val;
                }
            }
        }
    }
}

// ---------------------------------------------------------------------------
// host orchestration
// ---------------------------------------------------------------------------
struct CInfo {
    const float* w; const float* b;
    int O, C, ks, pad;
    half_t* A; int K, Mt, Kst;
};

static inline void elaunch_zero(float* d, int n, hipStream_t s)  { zero_k<<<(n + 255) / 256, 256, 0, s>>>(d, n); }
static inline void elaunch_copy(float* d, const float* a, int n, hipStream_t s) { copy_k<<<(n + 255) / 256, 256, 0, s>>>(d, a, n); }

extern "C" void kernel_launch(void* const* d_in, const int* in_sizes, int n_in,
                              void* d_out, int out_size, void* d_ws, size_t ws_size,
                              hipStream_t stream) {
    (void)in_sizes; (void)n_in; (void)ws_size;
    const float* fmap1 = (const float*)d_in[0];
    const float* fmap2 = (const float*)d_in[1];
    const float* net0_raw = (const float*)d_in[2];
    const float* net1_raw = (const float*)d_in[3];
    const float* cz0 = (const float*)d_in[4];
    const float* cr0 = (const float*)d_in[5];
    const float* cq0 = (const float*)d_in[6];
    const float* cz1 = (const float*)d_in[7];
    const float* cr1 = (const float*)d_in[8];
    const float* cq1 = (const float*)d_in[9];
    const float* projm = (const float*)d_in[10];
    const float* dmin = (const float*)d_in[11];
    const float* dmax = (const float*)d_in[12];
    float* out = (float*)d_out;

    int iters = out_size / (2 * HI * WI);
    if (iters < 1) iters = 1;

    // -------- workspace bump allocator (256B aligned) --------
    uintptr_t base = (uintptr_t)d_ws;
    size_t off = 0;
    auto alloc = [&](size_t bytes) -> void* {
        void* p = (void*)(base + off);
        off = (off + bytes + 255) & ~(size_t)255;
        return p;
    };
    float* PJ    = (float*)alloc(24 * sizeof(float));
    float* C0    = (float*)alloc((size_t)HW * D * sizeof(float));
    float* C1    = (float*)alloc((size_t)HW * 32 * sizeof(float));
    float* N0A   = (float*)alloc((size_t)HID * HW * sizeof(float));
    float* N0B   = (float*)alloc((size_t)HID * HW * sizeof(float));
    float* N1A   = (float*)alloc((size_t)HID * HW2 * sizeof(float));
    float* N1B   = (float*)alloc((size_t)HID * HW2 * sizeof(float));
    float* POOL  = (float*)alloc((size_t)HID * HW2 * sizeof(float));
    float* CAT16 = (float*)alloc((size_t)256 * HW2 * sizeof(float));
    float* T16Z  = (float*)alloc((size_t)HID * HW2 * sizeof(float));
    float* T16Q  = (float*)alloc((size_t)HID * HW2 * sizeof(float));
    float* CORRF = (float*)alloc((size_t)CORP * HW * sizeof(float));
    float* T1    = (float*)alloc((size_t)256 * HW * sizeof(float));
    float* T2    = (float*)alloc((size_t)128 * HW * sizeof(float));
    float* T3    = (float*)alloc((size_t)128 * HW * sizeof(float));
    float* CAT   = (float*)alloc((size_t)384 * HW * sizeof(float));
    float* MF    = (float*)alloc((size_t)128 * HW * sizeof(float));
    float* UP    = (float*)alloc((size_t)128 * HW * sizeof(float));
    float* DISP  = (float*)alloc((size_t)2 * HW * sizeof(float));
    float* DELTA = (float*)alloc((size_t)2 * HW * sizeof(float));
    float* MASK  = (float*)alloc((size_t)144 * HW * sizeof(float));
    float* DUP   = (float*)alloc((size_t)2 * HI * WI * sizeof(float));
    half_t* BF   = (half_t*)alloc((size_t)22 * 1024 * 1024);   // B fragments

    // -------- conv descriptors + weight packing --------
    enum { MEC1, MEC2, MEF1, MEF2, ME, G8Z, G8R, G8Q, G16Z, G16R, G16Q,
           FH1, FH2, MH1, MH2, NCONV };
    CInfo ci[NCONV];
    auto setc = [&](int id, int wi, int bi, int O, int C, int ks, int pad) {
        ci[id].w = (const float*)d_in[wi];
        ci[id].b = (const float*)d_in[bi];
        ci[id].O = O; ci[id].C = C; ci[id].ks = ks; ci[id].pad = pad;
        ci[id].K = C * ks * ks;
        ci[id].Mt = (O + 15) / 16;
        ci[id].Kst = (ci[id].K + 31) / 32;
        ci[id].A = (half_t*)alloc((size_t)ci[id].Mt * ci[id].Kst * 512 * sizeof(half_t));
    };
    setc(MEC1, 13, 14, 64, CORP, 1, 0);
    setc(MEC2, 15, 16, 64, 64, 3, 1);
    setc(MEF1, 17, 18, 64, 2, 7, 3);
    setc(MEF2, 19, 20, 64, 64, 3, 1);
    setc(ME,   21, 22, 126, 128, 3, 1);
    setc(G8Z,  23, 24, 128, 384, 3, 1);
    setc(G8R,  25, 26, 128, 384, 3, 1);
    setc(G8Q,  27, 28, 128, 384, 3, 1);
    setc(G16Z, 29, 30, 128, 256, 3, 1);
    setc(G16R, 31, 32, 128, 256, 3, 1);
    setc(G16Q, 33, 34, 128, 256, 3, 1);
    setc(FH1,  35, 36, 256, 128, 3, 1);
    setc(FH2,  37, 38, 2, 256, 3, 1);
    setc(MH1,  39, 40, 256, 128, 3, 1);
    setc(MH2,  41, 42, 144, 256, 1, 0);

    for (int i = 0; i < NCONV; ++i) {
        long tot = (long)ci[i].Mt * ci[i].Kst * 512;
        pack_w_k<<<(unsigned)((tot + 255) / 256), 256, 0, stream>>>(
            ci[i].w, ci[i].A, ci[i].O, ci[i].K, ci[i].Kst, tot);
    }

    auto conv = [&](int id, const float* src, int Hh, int Ww, float* dst, int act) {
        int Ntiles = (Hh * Ww) / 16;
        long btot = (long)Ntiles * ci[id].Kst * 512;
        im2col_k<<<(unsigned)((btot + 255) / 256), 256, 0, stream>>>(
            src, BF, ci[id].C, Hh, Ww, ci[id].ks, ci[id].pad, ci[id].Kst, btot);
        dim3 g(Ntiles / 4, (ci[id].Mt + 7) / 8);
        gemm_wmma_k<<<g, 256, 0, stream>>>(ci[id].A, BF, ci[id].b, dst,
                                           ci[id].O, Hh * Ww, ci[id].Kst, ci[id].Mt, act);
    };

    // -------- setup: projections, correlation pyramid, states --------
    compute_proj_k<<<1, 1, 0, stream>>>(projm, PJ);
    warp_corr_k<<<HW, 256, 0, stream>>>(fmap1, fmap2, PJ, dmin, dmax, C0);
    pyr1_k<<<(HW * 32 + 255) / 256, 256, 0, stream>>>(C0, C1, HW * 32);
    tanh_k<<<(HID * HW + 255) / 256, 256, 0, stream>>>(N0A, net0_raw, HID * HW);
    tanh_k<<<(HID * HW2 + 255) / 256, 256, 0, stream>>>(N1A, net1_raw, HID * HW2);
    elaunch_zero(DISP, 2 * HW, stream);

    float* n0 = N0A; float* n0n = N0B;
    float* n1 = N1A; float* n1n = N1B;

    for (int it = 0; it < iters; ++it) {
        // ---- g16 GRU: x = pool2x(net0); hx = [net1, x]
        pool2x_k<<<(HID * HW2 + 255) / 256, 256, 0, stream>>>(n0, POOL, HID * HW2);
        elaunch_copy(CAT16, n1, HID * HW2, stream);
        elaunch_copy(CAT16 + (size_t)HID * HW2, POOL, HID * HW2, stream);
        conv(G16Z, CAT16, H2, W2, T16Z, 0);
        conv(G16R, CAT16, H2, W2, T16Q, 0);
        gru_rh_k<<<(HID * HW2 + 255) / 256, 256, 0, stream>>>(T16Q, cr1, n1, CAT16, HID * HW2);
        conv(G16Q, CAT16, H2, W2, T16Q, 0);
        gru_comb_k<<<(HID * HW2 + 255) / 256, 256, 0, stream>>>(T16Z, cz1, T16Q, cq1, n1, n1n, HID * HW2);
        { float* t = n1; n1 = n1n; n1n = t; }

        // ---- motion encoder
        corr_lookup_k<<<(HW + 255) / 256, 256, 0, stream>>>(DISP, C0, C1, CORRF);
        conv(MEC1, CORRF, H, W, T1, 1);                  // 64ch
        conv(MEC2, T1, H, W, T2, 1);                     // cf -> T2
        conv(MEF1, DISP, H, W, T1, 1);                   // 64ch
        conv(MEF2, T1, H, W, T3, 1);                     // ff -> T3
        elaunch_copy(CAT, T2, 64 * HW, stream);
        elaunch_copy(CAT + (size_t)64 * HW, T3, 64 * HW, stream);
        conv(ME, CAT, H, W, T1, 1);                      // 126ch
        elaunch_copy(MF, T1, 126 * HW, stream);
        elaunch_copy(MF + (size_t)126 * HW, DISP, 2 * HW, stream);

        // ---- x0 = [mf, up(net1)]; g8 GRU on hx = [net0, mf, up]
        interp_up_k<<<(HID * HW + 255) / 256, 256, 0, stream>>>(n1, UP, HID * HW);
        elaunch_copy(CAT, n0, HID * HW, stream);
        elaunch_copy(CAT + (size_t)HID * HW, MF, HID * HW, stream);
        elaunch_copy(CAT + (size_t)2 * HID * HW, UP, HID * HW, stream);
        conv(G8Z, CAT, H, W, T2, 0);
        conv(G8R, CAT, H, W, T3, 0);
        gru_rh_k<<<(HID * HW + 255) / 256, 256, 0, stream>>>(T3, cr0, n0, CAT, HID * HW);
        conv(G8Q, CAT, H, W, T3, 0);
        gru_comb_k<<<(HID * HW + 255) / 256, 256, 0, stream>>>(T2, cz0, T3, cq0, n0, n0n, HID * HW);
        { float* t = n0; n0 = n0n; n0n = t; }

        // ---- heads
        conv(FH1, n0, H, W, T1, 1);
        conv(FH2, T1, H, W, DELTA, 0);
        add_inplace_k<<<(2 * HW + 255) / 256, 256, 0, stream>>>(DISP, DELTA, 2 * HW);
        conv(MH1, n0, H, W, T1, 1);
        conv(MH2, T1, H, W, MASK, 0);

        upsample_k<<<(2 * HI * WI + 255) / 256, 256, 0, stream>>>(DISP, MASK, DUP, 2 * HI * WI);
        depth_k<<<(2 * HI * WI + 255) / 256, 256, 0, stream>>>(
            DUP, dmin, dmax, out + (size_t)it * 2 * HI * WI, 2 * HI * WI);
    }
}
